// OrthogonalMatchingPursuitSecondVersion_84421877170534
// MI455X (gfx1250) — matrix operand order repaired and checked by reading, not compiled, not run on hardware
//
#include <hip/hip_runtime.h>

// Problem constants (from reference)
#define B_  64
#define L_  512
#define A_  4096
#define A1_ 4097
#define K_  16
#define MT_ 257   // atom tiles of 16 (covers 4112 >= 4097, padded with zeros)
#define NT_ 4     // batch tiles of 16

typedef __attribute__((ext_vector_type(16))) _Float16 v16h;
typedef __attribute__((ext_vector_type(8)))  float    v8f;

// ---------------------------------------------------------------------------
// Fragment layouts used here (wave32; contiguous k order, not the ISA table order):
//  A (16x32 f16): lane lo=lane%16 holds row M=lo; hi=lane/16;
//     half j (0..15): K = 16*hi + j
//  B (32x16 f16): lane holds column N=lane%16; K = 16*(lane/16) + j
//  C/D (16x16 f32): N=lane%16, M = v + 8*(lane/16)
// A/B fragments are pre-swizzled in memory so the GEMM loop loads 32B/lane.
// ---------------------------------------------------------------------------

// Build swizzled f16 A-fragments of D^T (shared dictionary, batch 0) + bias col.
__global__ __launch_bounds__(256)
void build_afrag_kernel(const float* __restrict__ dict0, _Float16* __restrict__ Afrag) {
    int e = blockIdx.x * 256 + threadIdx.x;          // element index
    if (e >= MT_ * 16 * 32 * 16) return;             // 2,105,344 halves
    int j     = e & 15;
    int lane  = (e >> 4) & 31;
    int k0    = (e >> 9) & 15;                       // 16 k-steps of 32
    int tileM = e >> 13;                             // 8192 halves per tile
    int v = j >> 1, p = j & 1, hi = lane >> 4;
    int k    = k0 * 32 + 16 * hi + j; // l index
    int atom = tileM * 16 + (lane & 15);
    float val = (atom < A_) ? dict0[(size_t)k * A_ + atom]
                            : (atom == A_ ? 1.0f : 0.0f);
    Afrag[e] = (_Float16)val;
}

// Initial residual = y: pack into B-fragments; zero argmax keys.
__global__ __launch_bounds__(256)
void init_kernel(const float* __restrict__ y, _Float16* __restrict__ Bfrag,
                 unsigned long long* __restrict__ keys) {
    int e = blockIdx.x * 256 + threadIdx.x;
    if (e < B_) keys[e] = 0ull;
    if (e >= NT_ * 16 * 32 * 16) return;             // 32,768 halves
    int j     = e & 15;
    int lane  = (e >> 4) & 31;
    int k0    = (e >> 9) & 15;
    int tileN = e >> 13;
    int n = tileN * 16 + (lane & 15);                // batch
    int l = k0 * 32 + 16 * (lane >> 4) + j;
    Bfrag[e] = (_Float16)y[(size_t)n * L_ + l];
}

// Per-accumulator |proj| argmax folded into packed 64-bit atomicMax keys.
__device__ inline void tile_argmax(v8f c, int tileM, int tileN, int lane,
                                   unsigned long long* __restrict__ keys) {
    const int lo = lane & 15, hi = lane >> 4;
    unsigned long long best = 0ull;
#pragma unroll
    for (int v = 0; v < 8; ++v) {
        int atom = tileM * 16 + v + 8 * hi;
        unsigned int ab = __float_as_uint(c[v]) & 0x7FFFFFFFu; // |proj| bits
        unsigned long long key =
            ((unsigned long long)ab << 32) | (unsigned int)(~atom); // smaller atom wins ties
        if (atom < A1_ && key > best) best = key;
    }
    // Combine the two lanes owning this batch column (lane lo and lo+16).
    unsigned int blo = (unsigned int)best, bhi = (unsigned int)(best >> 32);
    unsigned int olo = __shfl_xor(blo, 16, 32);
    unsigned int ohi = __shfl_xor(bhi, 16, 32);
    unsigned long long other = ((unsigned long long)ohi << 32) | olo;
    if (other > best) best = other;
    if (hi == 0) atomicMax(&keys[tileN * 16 + lo], best);
}

// proj = D^T * r on WMMA. One wave per atom-tile handles ALL 4 batch tiles:
// each A-fragment load feeds 4 WMMAs (4x less A traffic, 64 wmma per wave).
__global__ __launch_bounds__(32)
void proj_argmax_kernel(const _Float16* __restrict__ Afrag,
                        const _Float16* __restrict__ Bfrag,
                        unsigned long long* __restrict__ keys) {
    const int tileM = blockIdx.x;   // 0..256
    const int lane  = threadIdx.x;  // wave32, all lanes active (EXEC all 1s)

    const v16h* Av = (const v16h*)Afrag;
    const v16h* Bv = (const v16h*)Bfrag;

    v8f c0 = {}, c1 = {}, c2 = {}, c3 = {};
#pragma unroll
    for (int k0 = 0; k0 < 16; ++k0) {
        v16h a  = Av[((size_t)(tileM * 16 + k0)) * 32 + lane];
        v16h b0 = Bv[((size_t)(0 * 16 + k0)) * 32 + lane];
        v16h b1 = Bv[((size_t)(1 * 16 + k0)) * 32 + lane];
        v16h b2 = Bv[((size_t)(2 * 16 + k0)) * 32 + lane];
        v16h b3 = Bv[((size_t)(3 * 16 + k0)) * 32 + lane];
        c0 = __builtin_amdgcn_wmma_f32_16x16x32_f16(false, a, false, b0, (short)0, c0, false, false);
        c1 = __builtin_amdgcn_wmma_f32_16x16x32_f16(false, a, false, b1, (short)0, c1, false, false);
        c2 = __builtin_amdgcn_wmma_f32_16x16x32_f16(false, a, false, b2, (short)0, c2, false, false);
        c3 = __builtin_amdgcn_wmma_f32_16x16x32_f16(false, a, false, b3, (short)0, c3, false, false);
    }

    tile_argmax(c0, tileM, 0, lane, keys);
    tile_argmax(c1, tileM, 1, lane, keys);
    tile_argmax(c2, tileM, 2, lane, keys);
    tile_argmax(c3, tileM, 3, lane, keys);
}

// Serial Cholesky solve of SPD n x n (n <= 16) on one thread.
__device__ void chol_solve(float M[16][16], const float* rhs, float* sol, int n) {
    for (int k = 0; k < n; ++k) {
        float d = sqrtf(M[k][k]);
        M[k][k] = d;
        float inv = 1.0f / d;
        for (int i = k + 1; i < n; ++i) M[i][k] *= inv;
        for (int j = k + 1; j < n; ++j)
            for (int i = j; i < n; ++i) M[i][j] -= M[i][k] * M[j][k];
    }
    float tmp[16];
    for (int i = 0; i < n; ++i) {
        float s = rhs[i];
        for (int k = 0; k < i; ++k) s -= M[i][k] * tmp[k];
        tmp[i] = s / M[i][i];
    }
    for (int i = n - 1; i >= 0; --i) {
        float s = tmp[i];
        for (int k = i + 1; k < n; ++k) s -= M[k][i] * sol[k];
        sol[i] = s / M[i][i];
    }
}

// One block per batch: read argmax, gather column from shared dict, update
// Gram/WTy, Cholesky solve, recompute residual, repack B-fragments.
__global__ __launch_bounds__(256)
void update_kernel(const float* __restrict__ dict, const float* __restrict__ y,
                   float* __restrict__ Wsel, float* __restrict__ G,
                   float* __restrict__ cvec, unsigned long long* __restrict__ keys,
                   int* __restrict__ idxbuf, _Float16* __restrict__ Bfrag, int iter) {
    __shared__ float red[256];
    __shared__ float s_sol[16];
    __shared__ unsigned int s_atom;
    const int b = blockIdx.x, tid = threadIdx.x;

    if (tid == 0) {
        unsigned long long kk = keys[b];
        keys[b] = 0ull;                       // re-arm for next proj pass
        s_atom = ~(unsigned int)(kk & 0xFFFFFFFFull);
        idxbuf[b * K_ + iter] = (int)s_atom;
    }
    __syncthreads();
    const int atom = (int)s_atom;

    float* wcol = Wsel + ((size_t)b * K_ + iter) * L_;
    for (int l = tid; l < L_; l += 256) {
        float v = (atom < A_) ? dict[(size_t)l * A_ + atom]
                              : (atom == A_ ? 1.0f : 0.0f);
        wcol[l] = v;
    }
    __syncthreads();

    const float* yb = y + (size_t)b * L_;
    for (int j = 0; j <= iter + 1; ++j) {
        const float* other = (j <= iter) ? (Wsel + ((size_t)b * K_ + j) * L_) : yb;
        float p = 0.0f;
        for (int l = tid; l < L_; l += 256) p += wcol[l] * other[l];
        red[tid] = p;
        __syncthreads();
        for (int s = 128; s > 0; s >>= 1) {
            if (tid < s) red[tid] += red[tid + s];
            __syncthreads();
        }
        if (tid == 0) {
            if (j <= iter) {
                G[(size_t)b * 256 + iter * 16 + j] = red[0];
                G[(size_t)b * 256 + j * 16 + iter] = red[0];
            } else {
                cvec[b * 16 + iter] = red[0];
            }
        }
        __syncthreads();
    }

    if (tid == 0) {
        float Mloc[16][16], rhs[16];
        int n = iter + 1;
        for (int i = 0; i < n; ++i) {
            rhs[i] = cvec[b * 16 + i];
            for (int j = 0; j < n; ++j) Mloc[i][j] = G[(size_t)b * 256 + i * 16 + j];
        }
        chol_solve(Mloc, rhs, s_sol, n);
    }
    __syncthreads();

    // residual = y - W * sol, packed directly into B-fragment layout (f16)
    const int lo_b = b & 15, tileN = b >> 4;
    for (int l = tid; l < L_; l += 256) {
        float r = yb[l];
        for (int j = 0; j <= iter; ++j)
            r -= Wsel[((size_t)b * K_ + j) * L_ + l] * s_sol[j];
        int k0 = l >> 5, rem = l & 31, hi = rem >> 4, jj = rem & 15;
        int lane = lo_b + (hi << 4);
        Bfrag[(((size_t)(tileN * 16 + k0)) * 32 + lane) * 16 + jj] = (_Float16)r;
    }
}

// One block per batch: gather 16 atoms from the FULL per-batch dictionary,
// solve normal equations, write reconstruction X * coef.
__global__ __launch_bounds__(256)
void final_kernel(const float* __restrict__ dict, const float* __restrict__ y,
                  const int* __restrict__ idxbuf, float* __restrict__ out) {
    __shared__ float sel[16][L_];   // 32 KB LDS
    __shared__ float ata[16][16];
    __shared__ float aty[16];
    __shared__ float coef[16];
    const int b = blockIdx.x, tid = threadIdx.x;
    const float* dictb = dict + (size_t)b * L_ * A_;
    const float* yb = y + (size_t)b * L_;

    for (int k = 0; k < 16; ++k) {
        int atom = idxbuf[b * 16 + k];
        for (int l = tid; l < L_; l += 256)
            sel[k][l] = (atom < A_) ? dictb[(size_t)l * A_ + atom]
                                    : (atom == A_ ? 1.0f : 0.0f);
    }
    __syncthreads();

    {   // 256 threads == 16x16 Gram entries
        int k = tid >> 4, m = tid & 15;
        float s = 0.0f;
        for (int l = 0; l < L_; ++l) s += sel[k][l] * sel[m][l];
        ata[k][m] = s;
    }
    if (tid < 16) {
        float s = 0.0f;
        for (int l = 0; l < L_; ++l) s += sel[tid][l] * yb[l];
        aty[tid] = s;
    }
    __syncthreads();

    if (tid == 0) {
        float Mloc[16][16], rhs[16];
        for (int i = 0; i < 16; ++i) {
            rhs[i] = aty[i];
            for (int j = 0; j < 16; ++j) Mloc[i][j] = ata[i][j];
        }
        chol_solve(Mloc, rhs, coef, 16);
    }
    __syncthreads();

    for (int l = tid; l < L_; l += 256) {
        float s = 0.0f;
#pragma unroll
        for (int k = 0; k < 16; ++k) s += sel[k][l] * coef[k];
        out[(size_t)b * L_ + l] = s;
    }
}

extern "C" void kernel_launch(void* const* d_in, const int* in_sizes, int n_in,
                              void* d_out, int out_size, void* d_ws, size_t ws_size,
                              hipStream_t stream) {
    (void)in_sizes; (void)n_in; (void)out_size; (void)ws_size;
    const float* dict = (const float*)d_in[0];   // [B, L, A] f32
    const float* y    = (const float*)d_in[1];   // [B, L, 1] f32
    float* out        = (float*)d_out;           // [B, L, 1] f32

    // Workspace layout (all 256B aligned)
    char* ws = (char*)d_ws;
    _Float16*           Afrag  = (_Float16*)(ws);                  // 4,210,688 B
    _Float16*           Bfrag  = (_Float16*)(ws + 4210688);        //    65,536 B
    unsigned long long* keys   = (unsigned long long*)(ws + 4276224); //    512 B
    int*                idxbuf = (int*)(ws + 4276736);             //     4,096 B
    float*              Wsel   = (float*)(ws + 4280832);           // 2,097,152 B
    float*              G      = (float*)(ws + 6377984);           //    65,536 B
    float*              cvec   = (float*)(ws + 6443520);           //     4,096 B

    build_afrag_kernel<<<(MT_ * 16 * 32 * 16 + 255) / 256, 256, 0, stream>>>(dict, Afrag);
    init_kernel<<<(NT_ * 16 * 32 * 16 + 255) / 256, 256, 0, stream>>>(y, Bfrag, keys);

    for (int i = 0; i < K_; ++i) {
        proj_argmax_kernel<<<MT_, 32, 0, stream>>>(Afrag, Bfrag, keys);
        update_kernel<<<B_, 256, 0, stream>>>(dict, y, Wsel, G, cvec, keys, idxbuf, Bfrag, i);
    }
    final_kernel<<<B_, 256, 0, stream>>>(dict, y, idxbuf, out);
}
